// VQCodebook_34754875359472
// MI455X (gfx1250) — matrix operand order repaired and checked by reading, not compiled, not run on hardware
//
#include <hip/hip_runtime.h>

// ---------------------------------------------------------------------------
// VQ codebook lookup for MI455X (gfx1250, wave32, WMMA).
// N=16384 rows (16 images x 1024 positions), D=256 dims, K=8192 codes.
//
// dist' = ||c||^2 - 2 z.c  (||z||^2 constant per row, dropped for argmin)
// Cross term via 3-term split bf16 WMMA: zh*ch + zl*ch + zh*cl  (~2^-16 rel).
// ---------------------------------------------------------------------------

#define K_CODES 8192
#define DIMS    256
#define BATCH   16
#define HWSZ    1024
#define NROWS   (BATCH * HWSZ)        // 16384
#define NELEM   (BATCH * DIMS * HWSZ) // 4194304
#define LOSS_OFF  NELEM
#define CODES_OFF (NELEM + 1)

typedef __attribute__((ext_vector_type(16))) __bf16 v16bf;
typedef __attribute__((ext_vector_type(8)))  float  v8f;

// Packed codebook layout (written by vq_pack into the front of d_out, which is
// free scratch until vq_gather_loss overwrites it):
//   128 groups of 64 codes; per group: [hi plane][lo plane], each plane is
//   64 code-rows of 544 bytes (256 bf16 data + 16 bf16 pad).
//   544 B pitch => every 32B WMMA fragment read is 32B-aligned and code rows
//   are staggered by 8 banks (2-way conflicts max).
#define CPITCH_B   544                       // bytes per code row
#define PLANE_B    (64 * CPITCH_B)           // 34816 B
#define GROUP_B    (2 * PLANE_B)             // 69632 B
#define NGROUPS    (K_CODES / 64)            // 128
#define PLANE_E    (PLANE_B / 2)             // plane offset in bf16 elems
#define CPITCH_E   (CPITCH_B / 2)            // 272 bf16 elems

// workspace layout (bytes):
//   [0, 32768)        cnorm   (8192 f32)
//   [32768, 98304)    codes   (16384 i32)
//   [98304, 163840)   partial (16384 f32)

// ---------------------------------------------------------------------------
// Kernel 0: pack codebook -> bf16 hi/lo planes in WMMA-B tile layout.
// One block per code row, one thread per dim. Coalesced 2B stores.
// ---------------------------------------------------------------------------
__global__ __launch_bounds__(256) void vq_pack(const float* __restrict__ cw,
                                               char* __restrict__ packed) {
  const int code = blockIdx.x;
  const int d    = threadIdx.x;
  const float v  = cw[(size_t)code * DIMS + d];
  const __bf16 h = (__bf16)v;
  const __bf16 l = (__bf16)(v - (float)h);
  char* base = packed + (size_t)(code >> 6) * GROUP_B + (code & 63) * CPITCH_B + 2 * d;
  *(__bf16*)(base)           = h;
  *(__bf16*)(base + PLANE_B) = l;
}

// ---------------------------------------------------------------------------
// Kernel 1: per-code squared norms (exact fp32). One wave32 per code row.
// ---------------------------------------------------------------------------
__global__ __launch_bounds__(256) void vq_cnorm(const float* __restrict__ cw,
                                                float* __restrict__ cnorm) {
  const int lane = threadIdx.x & 31;
  const int row  = blockIdx.x * 8 + (threadIdx.x >> 5);
  const float* rp = cw + (size_t)row * DIMS;
  float s = 0.0f;
#pragma unroll
  for (int j = 0; j < DIMS / 32; ++j) {
    float v = rp[lane + j * 32];
    s += v * v;
  }
#pragma unroll
  for (int m = 16; m; m >>= 1) s += __shfl_xor(s, m, 32);
  if (lane == 0) cnorm[row] = s;
}

// ---------------------------------------------------------------------------
// Kernel 2: fused distance GEMM + argmin.
// Block = 128 threads = 4 waves; wave w owns 32 rows (two 16-row A blocks,
// register-stationary bf16 hi/lo in the WMMA A layout; ~256 A VGPRs).
// 64-code B tiles staged into LDS by linear copy of the prepacked image;
// each B fragment read feeds 6 WMMAs (two accumulator chains).
// ---------------------------------------------------------------------------
__global__ __launch_bounds__(128) void vq_dist_argmin(
    const float* __restrict__ z, const char* __restrict__ packed,
    const float* __restrict__ cnorm, int* __restrict__ codes) {
  extern __shared__ __align__(32) char smem[];
  float*  s_cn   = (float*)smem;             // 32768 B
  char*   s_tile = smem + 32768;             // GROUP_B bytes (hi plane, lo plane)

  const int tid  = threadIdx.x;
  const int lane = tid & 31;
  const int wave = tid >> 5;
  const int nlcl = lane & 15;            // B column / C column (N = lane & 15)
  const int kA   = (lane >> 4) << 3;     // A-layout K offset: 0 or 8
  const int kB   = (lane >> 4) << 4;     // contiguous-order K offset (A and B): 0 or 16

  // stage cnorm into LDS (coalesced)
  for (int i = tid; i < K_CODES; i += 128) s_cn[i] = cnorm[i];

  // ---- build A: 2 row-blocks x 16 rows x 256 dims, split bf16 hi+lo ----
  const int r0 = (blockIdx.x * 4 + wave) * 32;
  v16bf ah[2][8], al[2][8];
#pragma unroll
  for (int m = 0; m < 2; ++m) {
    const int row = r0 + m * 16 + nlcl;                     // M = lane & 15
    const float* zrow =
        z + (size_t)(row >> 10) * (DIMS * HWSZ) + (row & (HWSZ - 1));
#pragma unroll
    for (int s = 0; s < 8; ++s) {
#pragma unroll
      for (int j = 0; j < 16; ++j) {
        // element j of the A operand holds K = s*32 + map(j, lane half)
        int kk = s * 32 + kB + j;
        float v = zrow[(size_t)kk * HWSZ];
        __bf16 h = (__bf16)v;
        ah[m][s][j] = h;
        al[m][s][j] = (__bf16)(v - (float)h);
      }
    }
  }

  float mv[2][8];
  int   mi[2][8];
#pragma unroll
  for (int m = 0; m < 2; ++m)
#pragma unroll
    for (int r = 0; r < 8; ++r) { mv[m][r] = 3.0e38f; mi[m][r] = 0; }

  for (int g = 0; g < NGROUPS; ++g) {
    __syncthreads();
    // ---- linear copy of prepacked 64-code tile: 128 thr x 544 B ----
    {
      const char* src = packed + (size_t)g * GROUP_B + tid * CPITCH_B;
      char*       dst = s_tile + tid * CPITCH_B;
#pragma unroll
      for (int i = 0; i < CPITCH_B / 16; ++i)
        *(uint4*)(dst + i * 16) = *(const uint4*)(src + i * 16);
    }
    __syncthreads();

    // ---- 4 sub-tiles of 16 codes; 48 WMMAs per sub-tile per wave ----
#pragma unroll
    for (int sub = 0; sub < 4; ++sub) {
      v8f acc0 = {0.f, 0.f, 0.f, 0.f, 0.f, 0.f, 0.f, 0.f};
      v8f acc1 = {0.f, 0.f, 0.f, 0.f, 0.f, 0.f, 0.f, 0.f};
      const __bf16* bp =
          (const __bf16*)s_tile + (sub * 16 + nlcl) * CPITCH_E + kB;
#pragma unroll
      for (int s = 0; s < 8; ++s) {
        v16bf bh = *(const v16bf*)(bp + s * 32);
        v16bf bl = *(const v16bf*)(bp + PLANE_E + s * 32);
        acc0 = __builtin_amdgcn_wmma_f32_16x16x32_bf16(false, ah[0][s], false, bh,
                                                       (short)0, acc0, false, false);
        acc1 = __builtin_amdgcn_wmma_f32_16x16x32_bf16(false, ah[1][s], false, bh,
                                                       (short)0, acc1, false, false);
        acc0 = __builtin_amdgcn_wmma_f32_16x16x32_bf16(false, al[0][s], false, bh,
                                                       (short)0, acc0, false, false);
        acc1 = __builtin_amdgcn_wmma_f32_16x16x32_bf16(false, al[1][s], false, bh,
                                                       (short)0, acc1, false, false);
        acc0 = __builtin_amdgcn_wmma_f32_16x16x32_bf16(false, ah[0][s], false, bl,
                                                       (short)0, acc0, false, false);
        acc1 = __builtin_amdgcn_wmma_f32_16x16x32_bf16(false, ah[1][s], false, bl,
                                                       (short)0, acc1, false, false);
      }
      // epilogue: dist' = ||c||^2 - 2*dot ; running argmin per lane
      const int code = g * 64 + sub * 16 + nlcl;
      const float cn = s_cn[code];
#pragma unroll
      for (int r = 0; r < 8; ++r) {          // M = r + 8*(lane>=16)
        float d0 = cn - 2.0f * acc0[r];
        float d1 = cn - 2.0f * acc1[r];
        if (d0 < mv[0][r]) { mv[0][r] = d0; mi[0][r] = code; }
        if (d1 < mv[1][r]) { mv[1][r] = d1; mi[1][r] = code; }
      }
    }
  }

  // ---- cross-lane reduce over the 16 columns sharing each row ----
#pragma unroll
  for (int m = 0; m < 2; ++m) {
#pragma unroll
    for (int r = 0; r < 8; ++r) {
      float v = mv[m][r];
      int   i = mi[m][r];
#pragma unroll
      for (int x = 1; x < 16; x <<= 1) {
        float ov = __shfl_xor(v, x, 32);
        int   oi = __shfl_xor(i, x, 32);
        if (ov < v || (ov == v && oi < i)) { v = ov; i = oi; }
      }
      if (nlcl == 0) codes[r0 + m * 16 + ((lane >> 4) << 3) + r] = i;
    }
  }
}

// ---------------------------------------------------------------------------
// Kernel 3: coalesced gather of winning codes, z_q_st output, squared-error
// partials (deterministic per-block reduction), codes emitted as float.
// ---------------------------------------------------------------------------
__global__ __launch_bounds__(256) void vq_gather_loss(
    const float* __restrict__ z, const float* __restrict__ cw,
    const int* __restrict__ codes, float* __restrict__ out,
    float* __restrict__ partial) {
  __shared__ float red[8];
  const int e  = blockIdx.x * 256 + threadIdx.x;  // flat [B, D, H, W] index
  const int hw = e & (HWSZ - 1);
  const int d  = (e >> 10) & (DIMS - 1);
  const int b  = e >> 18;
  const int rowi = (b << 10) | hw;

  const int code = codes[rowi];
  const float q  = cw[(size_t)code * DIMS + d];
  const float ze = z[e];
  out[e] = q;                                     // z_q_st == z_q value-wise
  const float df = ze - q;
  float s = df * df;
#pragma unroll
  for (int m = 16; m; m >>= 1) s += __shfl_xor(s, m, 32);
  if ((threadIdx.x & 31) == 0) red[threadIdx.x >> 5] = s;
  __syncthreads();
  if (threadIdx.x == 0) {
    float t = 0.0f;
#pragma unroll
    for (int i = 0; i < 8; ++i) t += red[i];
    partial[blockIdx.x] = t;
  }
  if (d == 0) out[CODES_OFF + rowi] = (float)code;
}

// ---------------------------------------------------------------------------
// Kernel 4: deterministic fixed-order final loss reduction.
// loss = (0.25 + 1.0) * mean((z - q)^2)
// ---------------------------------------------------------------------------
__global__ __launch_bounds__(256) void vq_loss_reduce(
    const float* __restrict__ partial, float* __restrict__ out) {
  __shared__ float sm[256];
  float s = 0.0f;
  for (int i = threadIdx.x; i < NELEM / 256; i += 256) s += partial[i];
  sm[threadIdx.x] = s;
  __syncthreads();
  for (int st = 128; st; st >>= 1) {
    if (threadIdx.x < st) sm[threadIdx.x] += sm[threadIdx.x + st];
    __syncthreads();
  }
  if (threadIdx.x == 0) out[LOSS_OFF] = sm[0] * (1.25f / (float)NELEM);
}

// ---------------------------------------------------------------------------
extern "C" void kernel_launch(void* const* d_in, const int* in_sizes, int n_in,
                              void* d_out, int out_size, void* d_ws, size_t ws_size,
                              hipStream_t stream) {
  const float* z  = (const float*)d_in[0];  // [16, 256, 32, 32]
  const float* cw = (const float*)d_in[1];  // [8192, 256]
  float* out = (float*)d_out;

  float* cnorm   = (float*)d_ws;
  int*   codes   = (int*)((char*)d_ws + 32768);
  float* partial = (float*)((char*)d_ws + 98304);

  // Packed bf16 codebook lives in the front of d_out (8.5 MB of the 16.8 MB
  // output buffer) until vq_gather_loss overwrites the whole output. This
  // avoids assuming anything about ws_size beyond 160 KB.
  char* packed = (char*)d_out;

  // 0) pack codebook into WMMA-B-layout bf16 hi/lo planes
  vq_pack<<<K_CODES, 256, 0, stream>>>(cw, packed);

  // 1) code norms: 8 waves/block -> 1024 blocks cover 8192 codes
  vq_cnorm<<<K_CODES / 8, 256, 0, stream>>>(cw, cnorm);

  // 2) fused distance GEMM + argmin:
  //    128 WGs x 4 waves x 32 rows = 16384 rows; 64-code LDS tiles.
  //    dynamic LDS: 32768 (cnorm) + 69632 (tile) = 102400 B
  const size_t lds = 32768 + GROUP_B;
  vq_dist_argmin<<<NROWS / 128, 128, lds, stream>>>(z, packed, cnorm, codes);

  // 3) gather + outputs + per-block loss partials (overwrites all of d_out)
  vq_gather_loss<<<NELEM / 256, 256, 0, stream>>>(z, cw, codes, out, partial);

  // 4) deterministic final reduction into the loss slot
  vq_loss_reduce<<<1, 256, 0, stream>>>(partial, out);
}
